// LinearTransformer_57887569215610
// MI455X (gfx1250) — matrix operand order repaired and checked by reading, not compiled, not run on hardware
//
#include <hip/hip_runtime.h>
#include <math.h>

// ---------------- types ----------------
typedef __bf16 bf16_t;
typedef __attribute__((ext_vector_type(16))) bf16_t v16bf;
typedef __attribute__((ext_vector_type(8)))  bf16_t v8bf;
typedef __attribute__((ext_vector_type(8)))  float  v8f;

#define NB    8
#define DIM   1024
#define FF    4096
#define BATCH 4
#define SEQ   4096
#define MTOT  (BATCH*SEQ)     // 16384 rows

#define BM 128
#define BN 128
#define BK 32
#define LDSP 40               // LDS row pitch in bf16 (80B rows: 16B aligned, conflict-free b128)

#if defined(__gfx1250__) && __has_builtin(__builtin_amdgcn_global_load_async_to_lds_b128)
#define HAVE_ASYNC_LDS 1
typedef int async_v4i __attribute__((vector_size(16)));
typedef __attribute__((address_space(1))) async_v4i gv4i_t;
typedef __attribute__((address_space(3))) async_v4i lv4i_t;
#else
#define HAVE_ASYNC_LDS 0
#endif

__device__ __forceinline__ void wait_async_lds() {
#if HAVE_ASYNC_LDS
 #if __has_builtin(__builtin_amdgcn_s_wait_asynccnt)
    __builtin_amdgcn_s_wait_asynccnt(0);
 #else
    asm volatile("s_wait_asynccnt 0" ::: "memory");
 #endif
#endif
}

// EPI: 0 = none, 1 = fmap (elu+1), 2 = gelu(tanh approx)
template<int EPI>
__device__ __forceinline__ float activate(float v) {
    if (EPI == 1) {
        return v > 0.0f ? v + 1.0f : __expf(v);
    } else if (EPI == 2) {
        float v3 = v * v * v;
        return 0.5f * v * (1.0f + tanhf(0.79788456080286535588f * (v + 0.044715f * v3)));
    }
    return v;
}

// stage one 128x32 A tile + 128x32 B tile into LDS (async direct-to-LDS when available)
__device__ __forceinline__ void stage_tile(const bf16_t* __restrict__ A,
                                           const bf16_t* __restrict__ Wt,
                                           bf16_t* la, bf16_t* lb,
                                           int bm, int bn, int k0, int K, int tid)
{
    #pragma unroll
    for (int r = 0; r < 2; ++r) {
        int chunk = tid + r * 256;
        int row   = chunk >> 2;
        int cc    = chunk & 3;
        const bf16_t* ga = A  + (size_t)(bm + row) * K + k0 + cc * 8;
        const bf16_t* gb = Wt + (size_t)(bn + row) * K + k0 + cc * 8;
        bf16_t* pa = la + row * LDSP + cc * 8;
        bf16_t* pb = lb + row * LDSP + cc * 8;
#if HAVE_ASYNC_LDS
        __builtin_amdgcn_global_load_async_to_lds_b128((gv4i_t*)ga, (lv4i_t*)pa, 0, 0);
        __builtin_amdgcn_global_load_async_to_lds_b128((gv4i_t*)gb, (lv4i_t*)pb, 0, 0);
#else
        *(uint4*)pa = *(const uint4*)ga;
        *(uint4*)pb = *(const uint4*)gb;
#endif
    }
}

// ---------------- WMMA GEMM ----------------
// C[M,N] = act(A[M,K] @ Wt[N,K]^T + bias), A/Wt bf16 row-major; outputs selected at compile time.
template<int EPI, bool HASF, bool HASB>
__global__ __launch_bounds__(256)
void gemm_bf16_wmma(const bf16_t* __restrict__ A,
                    const bf16_t* __restrict__ Wt,
                    const float*  __restrict__ bias,
                    float*        __restrict__ outF,
                    bf16_t*       __restrict__ outB,
                    int M, int N, int K)
{
    __shared__ bf16_t lA[2][BM * LDSP];
    __shared__ bf16_t lB[2][BN * LDSP];

    const int tid   = threadIdx.x;
    const int bm    = blockIdx.y * BM;
    const int bn    = blockIdx.x * BN;
    const int wave  = tid >> 5;
    const int lane  = tid & 31;
    const int laneh = lane >> 4;   // 0/1
    const int lanel = lane & 15;
    const int wm0   = (wave >> 1) * 32;   // 4 waves along M
    const int wn0   = (wave & 1) * 64;    // 2 waves along N

    v8f acc[2][4] = {};

    stage_tile(A, Wt, &lA[0][0], &lB[0][0], bm, bn, 0, K, tid);

    const int NK = K / BK;
    for (int kt = 0; kt < NK; ++kt) {
        const int cur = kt & 1;
        wait_async_lds();
        __syncthreads();
        if (kt + 1 < NK)
            stage_tile(A, Wt, &lA[cur ^ 1][0], &lB[cur ^ 1][0],
                       bm, bn, (kt + 1) * BK, K, tid);

        v16bf afrag[2];
        v16bf bfrag[4];
        #pragma unroll
        for (int i = 0; i < 2; ++i) {
            int m = wm0 + i * 16 + lanel;
            v8bf lo = *(const v8bf*)(&lA[cur][m * LDSP + 8 * laneh]);
            v8bf hi = *(const v8bf*)(&lA[cur][m * LDSP + 16 + 8 * laneh]);
            afrag[i] = __builtin_shufflevector(lo, hi,
                        0,1,2,3,4,5,6,7,8,9,10,11,12,13,14,15);
        }
        #pragma unroll
        for (int j = 0; j < 4; ++j) {
            int n = wn0 + j * 16 + lanel;
            v8bf blo = *(const v8bf*)(&lB[cur][n * LDSP + 8 * laneh]);
            v8bf bhi = *(const v8bf*)(&lB[cur][n * LDSP + 16 + 8 * laneh]);
            bfrag[j] = __builtin_shufflevector(blo, bhi,
                        0,1,2,3,4,5,6,7,8,9,10,11,12,13,14,15);
        }
        #pragma unroll
        for (int i = 0; i < 2; ++i)
            #pragma unroll
            for (int j = 0; j < 4; ++j)
                acc[i][j] = __builtin_amdgcn_wmma_f32_16x16x32_bf16(
                                false, afrag[i], false, bfrag[j],
                                (short)0, acc[i][j], false, false);
    }

    // straight-line epilogue: bias + activation + compile-time selected stores
    #pragma unroll
    for (int j = 0; j < 4; ++j) {
        const int n  = bn + wn0 + j * 16 + lanel;
        const float bv = bias[n];
        #pragma unroll
        for (int i = 0; i < 2; ++i) {
            #pragma unroll
            for (int r = 0; r < 8; ++r) {
                int row = bm + wm0 + i * 16 + 8 * laneh + r;
                float v = activate<EPI>(acc[i][j][r] + bv);
                size_t off = (size_t)row * N + n;
                if (HASF) outF[off] = v;
                if (HASB) outB[off] = (bf16_t)v;
            }
        }
    }
}

// ---------------- transpose + fp32->bf16 weight conversion ----------------
// in: fp32 [K,N] row-major  ->  out: bf16 [N,K] row-major
__global__ __launch_bounds__(256)
void transpose_cvt(const float* __restrict__ in, bf16_t* __restrict__ out, int K, int N)
{
    __shared__ bf16_t tile[32][33];
    int tx = threadIdx.x & 31;
    int ty = threadIdx.x >> 5;       // 0..7
    int n0 = blockIdx.x * 32;
    int k0 = blockIdx.y * 32;
    #pragma unroll
    for (int r = 0; r < 4; ++r) {
        int k = k0 + ty + r * 8;
        tile[ty + r * 8][tx] = (bf16_t)in[(size_t)k * N + n0 + tx];
    }
    __syncthreads();
    #pragma unroll
    for (int r = 0; r < 4; ++r) {
        int n = n0 + ty + r * 8;
        out[(size_t)n * K + k0 + tx] = tile[tx][ty + r * 8];
    }
}

// ---------------- x init: fp32 copy + bf16 copy ----------------
__global__ __launch_bounds__(256)
void init_x(const float* __restrict__ x, float* __restrict__ xf, bf16_t* __restrict__ xb)
{
    size_t idx = (size_t)blockIdx.x * 256 + threadIdx.x;
    float v = x[idx];
    xf[idx] = v;
    xb[idx] = (bf16_t)v;
}

// ---------------- KV / Ksum reduction over sequence ----------------
__global__ __launch_bounds__(256)
void kv_reduce(const float* __restrict__ Kf, const float* __restrict__ Vf,
               float* __restrict__ KV, float* __restrict__ Ksum)
{
    int h = blockIdx.x * 256 + threadIdx.x;  // 0..DIM-1
    int b = blockIdx.y;
    const float* kp = Kf + (size_t)b * SEQ * DIM + h;
    const float* vp = Vf + (size_t)b * SEQ * DIM + h;
    float kv = 0.0f, ks = 0.0f;
    for (int s = 0; s < SEQ; ++s) {
        float kk = kp[(size_t)s * DIM];
        float vv = vp[(size_t)s * DIM];
        kv += kk * vv;
        ks += kk;
    }
    KV[b * DIM + h]   = kv;
    Ksum[b * DIM + h] = ks;
}

// ---------------- V = Q*KV / (Q*Ksum + eps), bf16 out ----------------
__global__ __launch_bounds__(256)
void vadj_kernel(const float* __restrict__ Qf, const float* __restrict__ KV,
                 const float* __restrict__ Ksum, bf16_t* __restrict__ out)
{
    size_t idx = (size_t)blockIdx.x * 256 + threadIdx.x;
    int h = (int)(idx & (DIM - 1));
    int b = (int)(idx >> 22);            // SEQ*DIM = 2^22
    float q = Qf[idx];
    float z = q * Ksum[b * DIM + h] + 1e-6f;
    out[idx] = (bf16_t)(q * KV[b * DIM + h] / z);
}

// ---------------- residual + LayerNorm (row = 1024) ----------------
__global__ __launch_bounds__(256)
void ln_residual(const float* __restrict__ t, const float* __restrict__ res,
                 const float* __restrict__ scale, const float* __restrict__ bias,
                 float* __restrict__ outF, bf16_t* __restrict__ outB)
{
    int row = blockIdx.x;
    const float* tr = t   + (size_t)row * DIM;
    const float* rr = res + (size_t)row * DIM;
    float vals[4];
    float s = 0.0f, sq = 0.0f;
    #pragma unroll
    for (int u = 0; u < 4; ++u) {
        float v = tr[threadIdx.x + u * 256] + rr[threadIdx.x + u * 256];
        vals[u] = v;
        s += v; sq += v * v;
    }
    #pragma unroll
    for (int o = 16; o > 0; o >>= 1) {
        s  += __shfl_down(s,  o, 32);
        sq += __shfl_down(sq, o, 32);
    }
    __shared__ float red[8][2];
    int wave = threadIdx.x >> 5, lane = threadIdx.x & 31;
    if (lane == 0) { red[wave][0] = s; red[wave][1] = sq; }
    __syncthreads();
    if (threadIdx.x == 0) {
        float S = 0.0f, SQ = 0.0f;
        #pragma unroll
        for (int w = 0; w < 8; ++w) { S += red[w][0]; SQ += red[w][1]; }
        red[0][0] = S; red[0][1] = SQ;
    }
    __syncthreads();
    float mean = red[0][0] * (1.0f / (float)DIM);
    float var  = red[0][1] * (1.0f / (float)DIM) - mean * mean;
    float rstd = rsqrtf(var + 1e-6f);
    #pragma unroll
    for (int u = 0; u < 4; ++u) {
        int h = threadIdx.x + u * 256;
        float y = (vals[u] - mean) * rstd * scale[h] + bias[h];
        size_t off = (size_t)row * DIM + h;
        outF[off] = y;
        outB[off] = (bf16_t)y;
    }
}

// ---------------- host launch ----------------
extern "C" void kernel_launch(void* const* d_in, const int* in_sizes, int n_in,
                              void* d_out, int out_size, void* d_ws, size_t ws_size,
                              hipStream_t stream)
{
    const float* x    = (const float*)d_in[0];
    const float* Wq   = (const float*)d_in[1];  const float* bq = (const float*)d_in[2];
    const float* Wk   = (const float*)d_in[3];  const float* bk = (const float*)d_in[4];
    const float* Wv   = (const float*)d_in[5];  const float* bv = (const float*)d_in[6];
    const float* Wo   = (const float*)d_in[7];  const float* bo = (const float*)d_in[8];
    const float* l1s  = (const float*)d_in[9];  const float* l1b = (const float*)d_in[10];
    const float* W1   = (const float*)d_in[11]; const float* b1 = (const float*)d_in[12];
    const float* W2   = (const float*)d_in[13]; const float* b2 = (const float*)d_in[14];
    const float* l2s  = (const float*)d_in[15]; const float* l2b = (const float*)d_in[16];

    // workspace carve-up
    char* ws = (char*)d_ws;
    size_t off = 0;
    const size_t perW = 12ull * 1024 * 1024;      // bf16 elems per block of weights
    bf16_t* Wbf = (bf16_t*)ws;                    off += (size_t)NB * perW * 2;
    float*  xf  = (float*)(ws + off);             off += (size_t)MTOT * DIM * 4;
    bf16_t* xb  = (bf16_t*)(ws + off);            off += (size_t)MTOT * DIM * 2;
    float*  Qf  = (float*)(ws + off);             off += (size_t)MTOT * DIM * 4;
    float*  Kf  = (float*)(ws + off);             off += (size_t)MTOT * DIM * 4;
    float*  Vf  = (float*)(ws + off);             off += (size_t)MTOT * DIM * 4;
    bf16_t* Vadj= (bf16_t*)(ws + off);            off += (size_t)MTOT * DIM * 2;
    float*  tmpF= (float*)(ws + off);             off += (size_t)MTOT * DIM * 4;
    bf16_t* hB  = (bf16_t*)(ws + off);            off += (size_t)MTOT * FF * 2;
    float*  KVb = (float*)(ws + off);             off += (size_t)BATCH * DIM * 4;
    float*  KsB = (float*)(ws + off);             off += (size_t)BATCH * DIM * 4;

    // convert + transpose all weights to bf16 [N,K]
    for (int i = 0; i < NB; ++i) {
        bf16_t* wb = Wbf + (size_t)i * perW;
        transpose_cvt<<<dim3(DIM/32, DIM/32), 256, 0, stream>>>(Wq + (size_t)i*DIM*DIM, wb + 0u,        DIM, DIM);
        transpose_cvt<<<dim3(DIM/32, DIM/32), 256, 0, stream>>>(Wk + (size_t)i*DIM*DIM, wb + 1048576u,  DIM, DIM);
        transpose_cvt<<<dim3(DIM/32, DIM/32), 256, 0, stream>>>(Wv + (size_t)i*DIM*DIM, wb + 2097152u,  DIM, DIM);
        transpose_cvt<<<dim3(DIM/32, DIM/32), 256, 0, stream>>>(Wo + (size_t)i*DIM*DIM, wb + 3145728u,  DIM, DIM);
        transpose_cvt<<<dim3(FF/32,  DIM/32), 256, 0, stream>>>(W1 + (size_t)i*DIM*FF,  wb + 4194304u,  DIM, FF);
        transpose_cvt<<<dim3(DIM/32, FF/32),  256, 0, stream>>>(W2 + (size_t)i*FF*DIM,  wb + 8388608u,  FF,  DIM);
    }

    init_x<<<(MTOT*DIM)/256, 256, 0, stream>>>(x, xf, xb);

    const dim3 gD(DIM/BN, MTOT/BM);   // N=1024 GEMMs
    const dim3 gF(FF/BN,  MTOT/BM);   // N=4096 GEMM

    for (int i = 0; i < NB; ++i) {
        bf16_t* wb = Wbf + (size_t)i * perW;
        // --- linear attention sublayer ---
        gemm_bf16_wmma<1, true, false><<<gD, 256, 0, stream>>>(xb, wb + 0u,       bq + i*DIM, Qf,  nullptr, MTOT, DIM, DIM);
        gemm_bf16_wmma<1, true, false><<<gD, 256, 0, stream>>>(xb, wb + 1048576u, bk + i*DIM, Kf,  nullptr, MTOT, DIM, DIM);
        gemm_bf16_wmma<0, true, false><<<gD, 256, 0, stream>>>(xb, wb + 2097152u, bv + i*DIM, Vf,  nullptr, MTOT, DIM, DIM);
        kv_reduce<<<dim3(DIM/256, BATCH), 256, 0, stream>>>(Kf, Vf, KVb, KsB);
        vadj_kernel<<<(MTOT*DIM)/256, 256, 0, stream>>>(Qf, KVb, KsB, Vadj);
        gemm_bf16_wmma<2, true, false><<<gD, 256, 0, stream>>>(Vadj, wb + 3145728u, bo + i*DIM, tmpF, nullptr, MTOT, DIM, DIM);
        ln_residual<<<MTOT, 256, 0, stream>>>(tmpF, xf, l1s + i*DIM, l1b + i*DIM, xf, xb);
        // --- feed-forward sublayer ---
        gemm_bf16_wmma<2, false, true><<<gF, 256, 0, stream>>>(xb, wb + 4194304u, b1 + i*FF,  nullptr, hB, MTOT, FF,  DIM);
        gemm_bf16_wmma<2, true, false><<<gD, 256, 0, stream>>>(hB, wb + 8388608u, b2 + i*DIM, tmpF, nullptr, MTOT, DIM, FF);
        float* xo = (i == NB - 1) ? (float*)d_out : xf;
        ln_residual<<<MTOT, 256, 0, stream>>>(tmpF, xf, l2s + i*DIM, l2b + i*DIM, xo, xb);
    }
}
